// TSE_17231408791823
// MI455X (gfx1250) — matrix operand order repaired and checked by reading, not compiled, not run on hardware
//
#include <hip/hip_runtime.h>

// ---------------- problem constants ----------------
#define BB    4
#define NTOK  2048
#define CIN   1024
#define C3    3072
#define HN    16
#define DD    64
#define MTOT  (BB * NTOK)              // 8192 flattened rows
#define SCALEF 0.08838834764831845f    // (2*64)^-0.5

// ---------------- WMMA plumbing ----------------
typedef __attribute__((ext_vector_type(16))) __bf16 v16bf;
typedef __attribute__((ext_vector_type(8)))  float  v8f;
typedef __attribute__((ext_vector_type(4)))  unsigned int v4u;
typedef __attribute__((ext_vector_type(8)))  int    v8i_t;
typedef __attribute__((ext_vector_type(4)))  int    v4i_t;

#if defined(__HIP_DEVICE_COMPILE__) && __has_builtin(__builtin_amdgcn_tensor_load_to_lds) && __has_builtin(__builtin_amdgcn_s_wait_tensorcnt)
#define HAVE_TDM 1
#else
#define HAVE_TDM 0
#endif

__device__ __forceinline__ unsigned short f2bf(float f) {
  __bf16 h = (__bf16)f;
  return __builtin_bit_cast(unsigned short, h);
}

union FragU { uint4 u[2]; v16bf v; };
union US8  { uint4 u; unsigned short s[8]; };

__device__ __forceinline__ v16bf load_frag(const unsigned short* p0,
                                           const unsigned short* p1) {
  FragU x;
  x.u[0] = *(const uint4*)p0;
  x.u[1] = *(const uint4*)p1;
  return x.v;
}

__device__ __forceinline__ v8f wmma_bf16(v16bf a, v16bf b, v8f c) {
  // D = A(16x32 bf16) x B(32x16 bf16) + C(16x16 f32)
  return __builtin_amdgcn_wmma_f32_16x16x32_bf16(
      false, a, false, b, (short)0, c, false, false);
}

#if HAVE_TDM
// Issue one TDM 2D tile load: bf16 elements, `rows` x `cols` tile from a
// row-major tensor with `stride` elements per row, into LDS at lptr.
// Descriptor bit layout per CDNA5 ISA ch.8 (D# groups 0/1; groups 2/3 NULL-ish).
// This toolchain exposes the 6-operand builtin:
//   (uint32x4 g0, int32x8 g1, int32x4, int32x4, int32x8, i32 cpol)
__device__ __forceinline__ void tdm_load_2d(const void* gptr, void* lptr,
                                            unsigned rows, unsigned cols,
                                            unsigned stride) {
  unsigned long long ga = (unsigned long long)gptr;
  unsigned lds = (unsigned)(unsigned long long)lptr;  // generic LDS ptr low 32b = LDS offset
  v4u g0;
  g0[0] = 1u;                                          // count=1 (valid), user mode
  g0[1] = lds;                                         // lds_addr [63:32]
  g0[2] = (unsigned)(ga & 0xFFFFFFFFu);                // global_addr [95:64]
  g0[3] = (unsigned)((ga >> 32) & 0x01FFFFFFu) | (2u << 30); // addr[56:32] | type=2
  v8i_t g1;
  g1[0] = 0x00010000;                 // workgroup_mask=0, data_size=1 (2B), no flags
  g1[1] = (int)(cols << 16);          // atomic_barrier_addr=0 | tensor_dim0 lo16
  g1[2] = (int)(rows << 16);          // tensor_dim0 hi16=0 | tensor_dim1 lo16
  g1[3] = (int)(cols << 16);          // tensor_dim1 hi16=0 | tile_dim0
  g1[4] = (int)rows;                  // tile_dim1 | tile_dim2=0
  g1[5] = (int)stride;                // tensor_dim0_stride lo32
  g1[6] = 0;                          // stride hi16 | tensor_dim1_stride lo16
  g1[7] = 0;
  v4i_t z4 = {0, 0, 0, 0};
  v8i_t z8 = {0, 0, 0, 0, 0, 0, 0, 0};
  __builtin_amdgcn_tensor_load_to_lds(g0, g1, z4, z4, z8, 0);
}
#endif

// ---------------- kernel 1a: tokens f32 -> bf16 ----------------
__global__ void cvt_tokens(const float* __restrict__ tokens,
                           unsigned short* __restrict__ a_bf) {
  const long TOKN4 = (long)MTOT * CIN / 4;
  long stride = (long)gridDim.x * blockDim.x;
  for (long i = (long)blockIdx.x * blockDim.x + threadIdx.x; i < TOKN4; i += stride) {
    float4 f = ((const float4*)tokens)[i];
    ushort4 o;
    o.x = f2bf(f.x); o.y = f2bf(f.y); o.z = f2bf(f.z); o.w = f2bf(f.w);
    ((ushort4*)a_bf)[i] = o;
  }
}

// ---------------- kernel 1b: W1 [1024][3072] -> W1^T bf16 [3072][1024] -----
// LDS 32x32 tile transpose: both global sides coalesced.
__global__ __launch_bounds__(256) void w1t_kernel(const float* __restrict__ W1,
                                                  unsigned short* __restrict__ w1t) {
  __shared__ unsigned short tile[32][33];
  int nb = blockIdx.x % (C3 / 32), kb = blockIdx.x / (C3 / 32);
  int n0 = nb * 32, k0 = kb * 32;
  int tx = threadIdx.x & 31, tg = threadIdx.x >> 5;  // 32 x 8
  for (int r = 0; r < 4; ++r) {
    int k = tg * 4 + r;
    tile[tx][k] = f2bf(W1[(long)(k0 + k) * C3 + n0 + tx]);
  }
  __syncthreads();
  for (int r = 0; r < 4; ++r) {
    int n = tg * 4 + r;
    w1t[(long)(n0 + n) * CIN + k0 + tx] = tile[n][tx];
  }
}

// ---------------- kernel 2: sentence-stream scalar bias ----------------
__global__ void sen_kernel(const float* __restrict__ sen,
                           const float* __restrict__ W2,
                           float* __restrict__ s2) {
  int b = blockIdx.x >> 4, h = blockIdx.x & 15, d = threadIdx.x; // 64 threads
  const float* sp = sen + (long)b * CIN;
  float q2 = 0.f, k2 = 0.f;
  for (int k = 0; k < CIN; ++k) {
    float sv = sp[k];
    const float* w = W2 + (long)k * 2048 + h * 64 + d;
    q2 = fmaf(sv, w[0],    q2);
    k2 = fmaf(sv, w[1024], k2);
  }
  __shared__ float red[64];
  red[d] = q2 * k2;
  __syncthreads();
  for (int st = 32; st > 0; st >>= 1) {
    if (d < st) red[d] += red[d + st];
    __syncthreads();
  }
  if (d == 0) s2[blockIdx.x] = red[0];
}

// ---------------- kernel 3: QKV = tokens @ W1, bf16 WMMA GEMM ----------------
// 256 threads (8 waves). Block tile 128(M) x 64(N), K-step 64, double-buffered
// LDS fed by the Tensor Data Mover (wave 0 issues, TENSORcnt pipelined).
__global__ __launch_bounds__(256) void gemm_qkv(
    const unsigned short* __restrict__ A,   // [8192][1024] bf16
    const unsigned short* __restrict__ BT,  // [3072][1024] bf16 (= W1^T)
    unsigned short* __restrict__ Cm) {      // [8192][3072] bf16
  __shared__ unsigned short sA[2][128][64];  // 32 KB
  __shared__ unsigned short sB[2][64][64];   // 16 KB
  int t = threadIdx.x, wave = t >> 5, lane = t & 31;
  int lrow = lane & 15, lhi = lane >> 4;
  int mb = blockIdx.x % (MTOT / 128);
  int nb = blockIdx.x / (MTOT / 128);
  int m0 = mb * 128, n0 = nb * 64;

  v8f acc[4] = {};

  auto stage = [&](int buf, int k0) {
#if HAVE_TDM
    if (wave == 0) {
      tdm_load_2d(A  + (long)m0 * CIN + k0, &sA[buf][0][0], 128, 64, CIN);
      tdm_load_2d(BT + (long)n0 * CIN + k0, &sB[buf][0][0],  64, 64, CIN);
    }
#else
    int row = t >> 1, seg = t & 1;   // A: 128 rows, 2 thr/row, 64B each
    const uint4* src = (const uint4*)(A + (long)(m0 + row) * CIN + k0 + seg * 32);
    uint4* dst = (uint4*)(&sA[buf][row][seg * 32]);
    dst[0] = src[0]; dst[1] = src[1]; dst[2] = src[2]; dst[3] = src[3];
    int n = t >> 2, sg = t & 3;      // B: 64 rows, 4 thr/row, 32B each
    const uint4* bs = (const uint4*)(BT + (long)(n0 + n) * CIN + k0 + sg * 16);
    uint4* bd = (uint4*)(&sB[buf][n][sg * 16]);
    bd[0] = bs[0]; bd[1] = bs[1];
#endif
  };

  stage(0, 0);
#if HAVE_TDM
  if (wave == 0) __builtin_amdgcn_s_wait_tensorcnt(0);
#endif
  __syncthreads();

  for (int it = 0; it < CIN / 64; ++it) {
    int cur = it & 1;
    if (it + 1 < CIN / 64) stage(1 - cur, (it + 1) * 64);  // overlap DMA w/ compute

    const unsigned short* ar = &sA[cur][wave * 16 + lrow][0];
    v16bf a0 = load_frag(ar + lhi * 16,      ar + lhi * 16 + 8);
    v16bf a1 = load_frag(ar + 32 + lhi * 16, ar + 32 + lhi * 16 + 8);
    for (int ns = 0; ns < 4; ++ns) {
      const unsigned short* br = &sB[cur][ns * 16 + lrow][0];
      v16bf b0 = load_frag(br + lhi * 16,      br + lhi * 16 + 8);
      acc[ns] = wmma_bf16(a0, b0, acc[ns]);
      v16bf b1 = load_frag(br + 32 + lhi * 16, br + 32 + lhi * 16 + 8);
      acc[ns] = wmma_bf16(a1, b1, acc[ns]);
    }
#if HAVE_TDM
    if (wave == 0 && it + 1 < CIN / 64) __builtin_amdgcn_s_wait_tensorcnt(0);
#endif
    __syncthreads();
  }

  for (int ns = 0; ns < 4; ++ns)
    for (int e = 0; e < 8; ++e) {
      int row = m0 + wave * 16 + e + 8 * lhi;
      int col = n0 + ns * 16 + lrow;
      Cm[(long)row * C3 + col] = f2bf(acc[ns][e]);
    }
}

// ---------------- kernel 4: flash attention ----------------
// 128 threads (4 waves). One block per (b, h, 64-query-row block).
// K tile staged via TDM; V tile transposed through LDS by the waves.
__global__ __launch_bounds__(128) void flash_kernel(
    const unsigned short* __restrict__ QKV,  // [8192][3072] bf16: [q|k|v]
    const float* __restrict__ s2,            // [4*16]
    const int* __restrict__ mask,            // [4][2048]
    float* __restrict__ out) {               // [4][2048][1024] f32
  __shared__ unsigned short sK[64][64];        // key tile   [j][d]
  __shared__ unsigned short sVT[64][64];       // value tile transposed [d][j]
  __shared__ unsigned short sP[4][16][64];     // per-wave P tile [row][j]
  __shared__ float sMadd[64];                  // mask addend per key

  int t = threadIdx.x, wave = t >> 5, lane = t & 31;
  int lrow = lane & 15, lhi = lane >> 4;
  int bid = blockIdx.x;
  int qb = bid & 31;
  int h  = (bid >> 5) & 15;
  int b  = bid >> 9;
  int q0 = qb * 64 + wave * 16;

  const unsigned short* qkb = QKV + (long)b * NTOK * C3;
  float s2v = s2[b * 16 + h];

  // Q fragments (A-layout), resident for the whole key loop
  v16bf qf[2];
  {
    const unsigned short* qr = qkb + (long)(q0 + lrow) * C3 + h * 64;
    for (int ks = 0; ks < 2; ++ks)
      qf[ks] = load_frag(qr + ks * 32 + lhi * 16,
                         qr + ks * 32 + lhi * 16 + 8);
  }

  v8f o[4] = {};
  float mrow[8], lacc[8];
  for (int e = 0; e < 8; ++e) { mrow[e] = -1e30f; lacc[e] = 0.f; }

  for (int j0 = 0; j0 < NTOK; j0 += 64) {
    __syncthreads();
#if HAVE_TDM
    if (wave == 0)
      tdm_load_2d(qkb + (long)j0 * C3 + 1024 + h * 64, &sK[0][0], 64, 64, C3);
#else
    {
      int jj = t >> 1, part = t & 1;
      const uint4* ksrc =
          (const uint4*)(qkb + (long)(j0 + jj) * C3 + 1024 + h * 64 + part * 32);
      uint4* kdst = (uint4*)(&sK[jj][part * 32]);
      kdst[0] = ksrc[0]; kdst[1] = ksrc[1]; kdst[2] = ksrc[2]; kdst[3] = ksrc[3];
    }
#endif
    {
      int jj = t >> 1, part = t & 1;
      // stage V transposed: sVT[d][j] = V[j][d]
      const unsigned short* vsrc =
          qkb + (long)(j0 + jj) * C3 + 2048 + h * 64 + part * 32;
      for (int c = 0; c < 4; ++c) {
        US8 v; v.u = *(const uint4*)(vsrc + c * 8);
        int dbase = part * 32 + c * 8;
        for (int e = 0; e < 8; ++e) sVT[dbase + e][jj] = v.s[e];
      }
      if (t < 64)
        sMadd[t] = -10000.0f * (1.0f - (float)mask[(long)b * NTOK + j0 + t]);
      if (j0 + 64 < NTOK)   // prefetch next V tile row into L2/L0
        __builtin_prefetch(qkb + (long)(j0 + 64 + jj) * C3 + 2048 + h * 64, 0, 3);
    }
#if HAVE_TDM
    if (wave == 0) __builtin_amdgcn_s_wait_tensorcnt(0);
#endif
    __syncthreads();

    // ---- S = SCALE*(Q K^T + s2) + mask ----
    v8f sf[4];
    for (int ns = 0; ns < 4; ++ns) {
      v8f acc = {};
      for (int ks = 0; ks < 2; ++ks) {
        v16bf bk = load_frag(&sK[ns * 16 + lrow][ks * 32 + lhi * 16],
                             &sK[ns * 16 + lrow][ks * 32 + lhi * 16 + 8]);
        acc = wmma_bf16(qf[ks], bk, acc);
      }
      float madd = sMadd[ns * 16 + lrow];
      for (int e = 0; e < 8; ++e)
        sf[ns][e] = SCALEF * (acc[e] + s2v) + madd;
    }

    // ---- online softmax (rows in fragment element e, half-wave lhi) ----
    float mn[8];
    for (int e = 0; e < 8; ++e)
      mn[e] = fmaxf(fmaxf(sf[0][e], sf[1][e]), fmaxf(sf[2][e], sf[3][e]));
    for (int msk = 1; msk < 16; msk <<= 1)
      for (int e = 0; e < 8; ++e)
        mn[e] = fmaxf(mn[e], __shfl_xor(mn[e], msk, 32));

    float al[8];
    for (int e = 0; e < 8; ++e) {
      float mne = fmaxf(mrow[e], mn[e]);
      al[e] = __expf(mrow[e] - mne);
      mrow[e] = mne;
    }
    for (int ds = 0; ds < 4; ++ds)
      for (int e = 0; e < 8; ++e) o[ds][e] *= al[e];

    float ps[8] = {0.f, 0.f, 0.f, 0.f, 0.f, 0.f, 0.f, 0.f};
    for (int ns = 0; ns < 4; ++ns)
      for (int e = 0; e < 8; ++e) {
        float p = __expf(sf[ns][e] - mrow[e]);
        ps[e] += p;
        sP[wave][e + 8 * lhi][ns * 16 + lrow] = f2bf(p);
      }
    for (int msk = 1; msk < 16; msk <<= 1)
      for (int e = 0; e < 8; ++e) ps[e] += __shfl_xor(ps[e], msk, 32);
    for (int e = 0; e < 8; ++e) lacc[e] = lacc[e] * al[e] + ps[e];

    // ---- O += P V ----
    v16bf pf[2];
    for (int ks = 0; ks < 2; ++ks)
      pf[ks] = load_frag(&sP[wave][lrow][ks * 32 + lhi * 16],
                         &sP[wave][lrow][ks * 32 + lhi * 16 + 8]);
    for (int ds = 0; ds < 4; ++ds)
      for (int ks = 0; ks < 2; ++ks) {
        v16bf bv = load_frag(&sVT[ds * 16 + lrow][ks * 32 + lhi * 16],
                             &sVT[ds * 16 + lrow][ks * 32 + lhi * 16 + 8]);
        o[ds] = wmma_bf16(pf[ks], bv, o[ds]);
      }
  }

  // ---- normalize and write out [b, tok, h*64 + d] ----
  for (int ds = 0; ds < 4; ++ds)
    for (int e = 0; e < 8; ++e) {
      int tok = q0 + e + 8 * lhi;
      int col = h * 64 + ds * 16 + lrow;
      out[((long)b * NTOK + tok) * (HN * DD) + col] = o[ds][e] / lacc[e];
    }
}

// ---------------- launcher ----------------
extern "C" void kernel_launch(void* const* d_in, const int* in_sizes, int n_in,
                              void* d_out, int out_size, void* d_ws, size_t ws_size,
                              hipStream_t stream) {
  const float* tokens = (const float*)d_in[0];
  const float* sen    = (const float*)d_in[1];
  const float* W1     = (const float*)d_in[2];
  const float* W2     = (const float*)d_in[3];
  const int*   mask   = (const int*)d_in[4];
  float* out = (float*)d_out;

  unsigned short* a_bf = (unsigned short*)d_ws;                       // 16 MB
  unsigned short* w1t  = a_bf + (size_t)MTOT * CIN;                   // 6 MB
  unsigned short* qkv  = w1t + (size_t)C3 * CIN;                      // 48 MB
  float* s2 = (float*)(qkv + (size_t)MTOT * C3);                      // 256 B

  cvt_tokens<<<4096, 256, 0, stream>>>(tokens, a_bf);
  w1t_kernel<<<(C3 / 32) * (CIN / 32), 256, 0, stream>>>(W1, w1t);
  sen_kernel<<<BB * HN, 64, 0, stream>>>(sen, W2, s2);
  gemm_qkv<<<(MTOT / 128) * (C3 / 64), 256, 0, stream>>>(a_bf, w1t, qkv);
  flash_kernel<<<BB * HN * (NTOK / 64), 128, 0, stream>>>(qkv, s2, mask, out);
}
